// DF_RNN_64871186038888
// MI455X (gfx1250) — matrix operand order repaired and checked by reading, not compiled, not run on hardware
//
#include <hip/hip_runtime.h>
#include <hip/hip_bf16.h>

// Problem constants (from reference): S,T,I,H = 50,2048,64,256
#define SS 50
#define TT 2048
#define II 64
#define HH 256
#define TC 64                 // timestep chunk held in LDS
#define NCHUNK (TT / TC)

#if __has_builtin(__builtin_amdgcn_global_load_async_to_lds_b32)
#define HAVE_ASYNC_LDS 1
#else
#define HAVE_ASYNC_LDS 0
#endif

#if __has_builtin(__builtin_amdgcn_tanhf)
#define FAST_TANH(x) __builtin_amdgcn_tanhf(x)
#elif __has_builtin(__builtin_amdgcn_tanh_f32)
#define FAST_TANH(x) __builtin_amdgcn_tanh_f32(x)
#else
#define FAST_TANH(x) tanhf(x)
#endif

typedef __attribute__((ext_vector_type(16))) _Float16 v16h;
typedef __attribute__((ext_vector_type(8)))  float    v8f;

#if HAVE_ASYNC_LDS
typedef __attribute__((address_space(1))) int gint_t;   // global int
typedef __attribute__((address_space(3))) int lint_t;   // LDS int
#endif

union V16H {
    v16h     v;
    uint32_t u[8];
    _Float16 h[16];
};

union H8 {
    _Float16 h[8];
    uint4    u4;
};

// WMMA f16 fragment layouts as used here (contiguous k, not ISA table order), wave32:
//  A (16x32): lane = g*16 + m; element j=2d+p -> K = g*16 + 2d + p
//  B (32x16): lane = g*16 + n; element j=2d+p -> K = g*16 + 2d + p
//  C/D (16x16 f32): col n = lane&15, row m = r + 8*(lane>>4)
// Fragments stored in LDS pre-packed as [frag][lane][16 halves] so each operand
// build is two contiguous ds_load_b128 per lane.

__device__ __forceinline__ int sx_pack_off(int r, int k) {   // A-side (x)
    const int m  = r & 15, mt = r >> 4;
    const int kt = k >> 5, kr = k & 31;
    const int g  = kr >> 4;
    const int dd = (kr & 15) >> 1;
    const int p  = kr & 1;
    return (((mt * 2 + kt) * 32) + (g * 16 + m)) * 16 + dd * 2 + p;
}

__device__ __forceinline__ int wih_pack_off(int n, int k) {  // B-side (Wih^T)
    const int nt = n >> 4, nc = n & 15;
    const int kt = k >> 5, kr = k & 31;
    const int g  = kr >> 4;
    const int d  = (kr & 15) >> 1;
    const int p  = kr & 1;
    return (((nt * 2 + kt) * 32) + (g * 16 + nc)) * 16 + d * 2 + p;
}

__device__ __forceinline__ void wait_async_all() {
#if HAVE_ASYNC_LDS
#if __has_builtin(__builtin_amdgcn_s_wait_asynccnt)
    __builtin_amdgcn_s_wait_asynccnt(0);
#else
    asm volatile("s_wait_asynccnt 0" ::: "memory");
#endif
#endif
}

__global__ void __launch_bounds__(256)
rnn_fused_kernel(const float* __restrict__ in0,    // input_data  S*T*65
                 const float* __restrict__ hid0,   // hidden      S*H
                 const float* __restrict__ fixedE, // fixed_eff   S*T
                 const float* __restrict__ Wih,    // S*H*I
                 const float* __restrict__ Whh,    // S*H*H
                 const float* __restrict__ bih,    // S*H
                 const float* __restrict__ bhh,    // S*H
                 const float* __restrict__ Wlin,   // S*1*H
                 const float* __restrict__ blin,   // S*1
                 float* __restrict__ sigma_out,    // d_out + 1  (S*T)
                 float* __restrict__ loglik_ws)    // d_ws, S floats
{
    __shared__ _Float16 sWhh[HH * HH];    // 128 KB  [m][k] (A-row layout)
    __shared__ _Float16 sWihP[II * HH];   //  32 KB  pre-packed B fragments
    __shared__ float    sU[TC * HH];      //  64 KB  pre-activations chunk
    __shared__ _Float16 sxP[TC * II];     //   8 KB  pre-packed A fragments
#if HAVE_ASYNC_LDS
    __shared__ float    sxRaw[2][TC * II];//  32 KB  async double buffer
#endif
    __shared__ float    sz[TC];           // z chunk
    __shared__ float    sf[TC];           // fixed_effects chunk
    __shared__ float    hbuf[2][HH];
    __shared__ _Float16 hbuf16[2][HH];
    __shared__ float    sbias[HH];
    __shared__ float    spart[TC * 8];    // per-step per-wave sigma partials

    const int s    = blockIdx.x;
    const int tid  = threadIdx.x;
    const int lane = tid & 31;
    const int w    = tid >> 5;      // wave id 0..7
    const int g    = lane >> 4;     // half-wave
    const int nc   = lane & 15;     // col (B/C/D) / row (A) within tile

    const float* x_s = in0 + (size_t)s * TT * (II + 1);

    // ---- Stage weights into LDS (global f32 -> LDS f16), once per series ----
    const float* Whh_s = Whh + (size_t)s * HH * HH;
    for (int e = tid; e < HH * HH; e += 256)
        sWhh[e] = (_Float16)Whh_s[e];

    const float* Wih_s = Wih + (size_t)s * HH * II;
    for (int e = tid; e < HH * II; e += 256) {
        int n = e >> 6;
        int k = e & (II - 1);
        sWihP[wih_pack_off(n, k)] = (_Float16)Wih_s[e];
    }
    for (int j = tid; j < HH; j += 256) {
        sbias[j] = bih[s * HH + j] + bhh[s * HH + j];
        float h0 = hid0[s * HH + j];
        hbuf[0][j]   = h0;
        hbuf16[0][j] = (_Float16)h0;
    }
    const float wlin_r = Wlin[s * HH + tid];
    const float blin_s = blin[s];

#if HAVE_ASYNC_LDS
    // kick off async fetch of chunk 0's x into the first raw buffer
    for (int i = 0; i < (TC * II) / 256; ++i) {
        const int e = tid + i * 256;
        const int r = e >> 6, c = e & (II - 1);
        const float* gp = &x_s[(size_t)r * (II + 1) + 1 + c];
        __builtin_amdgcn_global_load_async_to_lds_b32(
            (gint_t*)gp, (lint_t*)&sxRaw[0][e], 0, 0);
    }
#endif
    __syncthreads();

    // ---- Preload this wave's W_hh strip as resident WMMA A fragments ----
    // wave w owns rows [32w, 32w+32): M-tiles {2w, 2w+1}, 8 K-tiles each.
    V16H Ahh[2][8];
#pragma unroll
    for (int mi = 0; mi < 2; ++mi) {
        const int mg = (2 * w + mi) * 16 + nc;
#pragma unroll
        for (int kt = 0; kt < 8; ++kt) {
#pragma unroll
            for (int d = 0; d < 8; ++d) {
                const int kb = kt * 32 + g * 16 + d * 2;
                Ahh[mi][kt].u[d] = *(const uint32_t*)&sWhh[mg * HH + kb];
            }
        }
    }

    float loglik = 0.0f;
    int tpar = 0;
    const int hb0 = (2 * w + 0) * 16 + 8 * g;   // base row, M-tile 0
    const int hb1 = (2 * w + 1) * 16 + 8 * g;   // base row, M-tile 1

    for (int ch = 0; ch < NCHUNK; ++ch) {
        const int t0 = ch * TC;

        wait_async_all();          // own wave's async writes landed
        __syncthreads();           // everyone's writes visible; prev chunk done

        // -- convert/stage x chunk into packed A fragments; stage z/fixed --
        for (int i = 0; i < (TC * II) / 256; ++i) {
            const int e = tid + i * 256;
            const int r = e >> 6, c = e & (II - 1);
#if HAVE_ASYNC_LDS
            const float xv = sxRaw[ch & 1][e];
#else
            const float xv = x_s[(size_t)(t0 + r) * (II + 1) + 1 + c];
#endif
            sxP[sx_pack_off(r, c)] = (_Float16)xv;
        }
        if (tid < TC) {
            sz[tid] = x_s[(size_t)(t0 + tid) * (II + 1)];     // input_data[s,t,0]
            sf[tid] = fixedE[(size_t)s * TT + t0 + tid];
        }
        __syncthreads();

        // -- launch async fetch of the NEXT chunk (overlaps with U + recurrence)
        if (ch + 1 < NCHUNK) {
#if HAVE_ASYNC_LDS
            for (int i = 0; i < (TC * II) / 256; ++i) {
                const int e = tid + i * 256;
                const int r = e >> 6, c = e & (II - 1);
                const float* gp = &x_s[(size_t)(t0 + TC + r) * (II + 1) + 1 + c];
                __builtin_amdgcn_global_load_async_to_lds_b32(
                    (gint_t*)gp, (lint_t*)&sxRaw[(ch + 1) & 1][e], 0, 0);
            }
#else
            const float* nx = &x_s[(size_t)(t0 + TC) * (II + 1)];
            __builtin_prefetch(nx + (size_t)tid * 16, 0, 1);
#endif
        }

        // -- U = x @ Wih^T + (b_ih + b_hh): WMMA GEMM over the chunk --
#pragma unroll 1
        for (int q = 0; q < 8; ++q) {
            const int tile = w + q * 8;
            const int nt = tile & 15;     // hidden tile
            const int mt = tile >> 4;     // time tile
            v8f acc = {};
#pragma unroll
            for (int kt = 0; kt < 2; ++kt) {
                V16H a, b;
                const uint32_t* ap =
                    (const uint32_t*)&sxP[(((mt * 2 + kt) * 32) + lane) * 16];
                const uint32_t* bp =
                    (const uint32_t*)&sWihP[(((nt * 2 + kt) * 32) + lane) * 16];
#pragma unroll
                for (int d = 0; d < 8; ++d) { a.u[d] = ap[d]; b.u[d] = bp[d]; }
                acc = __builtin_amdgcn_wmma_f32_16x16x32_f16(
                    false, a.v, false, b.v, (short)0, acc, false, false);
            }
            const int n_g = nt * 16 + nc;
            const float bb = sbias[n_g];
#pragma unroll
            for (int r = 0; r < 8; ++r) {
                const int m = mt * 16 + r + 8 * g;
                sU[m * HH + n_g] = acc[r] + bb;
            }
        }
        __syncthreads();

        // -- serial recurrence over the chunk: ONE barrier per step --
        for (int tl = 0; tl < TC; ++tl) {
            const int cur = tpar, nxt = tpar ^ 1;

            // U pre-activations issued early (DS is in-order; hidden by XDL)
            float uu0[8], uu1[8];
            {
                const float4* u0p = (const float4*)&sU[tl * HH + hb0];
                const float4* u1p = (const float4*)&sU[tl * HH + hb1];
                ((float4*)uu0)[0] = u0p[0]; ((float4*)uu0)[1] = u0p[1];
                ((float4*)uu1)[0] = u1p[0]; ((float4*)uu1)[1] = u1p[1];
            }

            // 4 independent accumulator chains of depth 4 keep the XDL pipe full
            v8f c0a = {}, c0b = {}, c1a = {}, c1b = {};
#pragma unroll
            for (int kt = 0; kt < 8; kt += 2) {
                V16H b0, b1;
                const uint32_t* h0p =
                    (const uint32_t*)&hbuf16[cur][(kt + 0) * 32 + g * 16];
                const uint32_t* h1p =
                    (const uint32_t*)&hbuf16[cur][(kt + 1) * 32 + g * 16];
#pragma unroll
                for (int d = 0; d < 8; ++d) { b0.u[d] = h0p[d]; b1.u[d] = h1p[d]; }
                c0a = __builtin_amdgcn_wmma_f32_16x16x32_f16(
                    false, Ahh[0][kt].v,     false, b0.v, (short)0, c0a, false, false);
                c1a = __builtin_amdgcn_wmma_f32_16x16x32_f16(
                    false, Ahh[1][kt].v,     false, b0.v, (short)0, c1a, false, false);
                c0b = __builtin_amdgcn_wmma_f32_16x16x32_f16(
                    false, Ahh[0][kt + 1].v, false, b1.v, (short)0, c0b, false, false);
                c1b = __builtin_amdgcn_wmma_f32_16x16x32_f16(
                    false, Ahh[1][kt + 1].v, false, b1.v, (short)0, c1b, false, false);
            }

            float hv0[8], hv1[8];
#pragma unroll
            for (int r = 0; r < 8; ++r) {
                hv0[r] = FAST_TANH(c0a[r] + c0b[r] + uu0[r]);
                hv1[r] = FAST_TANH(c1a[r] + c1b[r] + uu1[r]);
            }
            // commit (all 16 cols identical; only n==0 lanes store, vectorized)
            if (nc == 0) {
                H8 p0, p1;
#pragma unroll
                for (int r = 0; r < 8; ++r) {
                    p0.h[r] = (_Float16)hv0[r];
                    p1.h[r] = (_Float16)hv1[r];
                }
                *(uint4*)&hbuf16[nxt][hb0] = p0.u4;
                *(uint4*)&hbuf16[nxt][hb1] = p1.u4;
                float4* f0 = (float4*)&hbuf[nxt][hb0];
                float4* f1 = (float4*)&hbuf[nxt][hb1];
                f0[0] = make_float4(hv0[0], hv0[1], hv0[2], hv0[3]);
                f0[1] = make_float4(hv0[4], hv0[5], hv0[6], hv0[7]);
                f1[0] = make_float4(hv1[0], hv1[1], hv1[2], hv1[3]);
                f1[1] = make_float4(hv1[4], hv1[5], hv1[6], hv1[7]);
            }
            __syncthreads();

            // per-wave sigma partial (consumed once per chunk, no 2nd barrier)
            float p = hbuf[nxt][tid] * wlin_r;
#pragma unroll
            for (int off = 16; off > 0; off >>= 1)
                p += __shfl_down(p, off, 32);
            if (lane == 0) spart[tl * 8 + w] = p;

            tpar ^= 1;
        }
        __syncthreads();

        // -- chunk epilogue: sigma + loglik, 64 threads in parallel --
        if (tid < TC) {
            float sum = blin_s;
#pragma unroll
            for (int i = 0; i < 8; ++i) sum += spart[tid * 8 + i];
            const float sg = fabsf(sum);
            sigma_out[(size_t)s * TT + t0 + tid] = sg;
            const float df = sz[tid] - sf[tid];
            loglik += -(df * df) / (2.0f * sg * sg);
        }
    }

    // ---- deterministic per-series loglik reduction (once) ----
    __syncthreads();
    if (tid < TC) spart[tid] = loglik;
    __syncthreads();
    if (tid == 0) {
        float acc = 0.0f;
        for (int i = 0; i < TC; ++i) acc += spart[i];
        loglik_ws[s] = acc;
    }
}

// Deterministic final reduction of per-series log-likelihoods.
__global__ void reduce_loglik_kernel(const float* __restrict__ ws,
                                     float* __restrict__ out)
{
    if (threadIdx.x == 0) {
        float acc = 0.0f;
        for (int i = 0; i < SS; ++i) acc += ws[i];
        out[0] = acc;
    }
}

extern "C" void kernel_launch(void* const* d_in, const int* in_sizes, int n_in,
                              void* d_out, int out_size, void* d_ws, size_t ws_size,
                              hipStream_t stream) {
    const float* in0    = (const float*)d_in[0];  // input_data
    const float* hid0   = (const float*)d_in[1];  // hidden
    const float* fixedE = (const float*)d_in[2];  // fixed_effects
    const float* Wih    = (const float*)d_in[3];
    const float* Whh    = (const float*)d_in[4];
    const float* bih    = (const float*)d_in[5];
    const float* bhh    = (const float*)d_in[6];
    const float* Wlin   = (const float*)d_in[7];
    const float* blin   = (const float*)d_in[8];
    // d_in[9] = gaussian_likelihood (1), d_in[10] = prediction (0) -> Gaussian path

    float* out = (float*)d_out;          // out[0] = loglik, out[1..] = sigma (S*T)
    float* ws  = (float*)d_ws;           // SS floats of scratch

    rnn_fused_kernel<<<dim3(SS), dim3(256), 0, stream>>>(
        in0, hid0, fixedE, Wih, Whh, bih, bhh, Wlin, blin, out + 1, ws);
    reduce_loglik_kernel<<<dim3(1), dim3(32), 0, stream>>>(ws, out);
}
